// AttentionCritic_70360154243251
// MI455X (gfx1250) — matrix operand order repaired and checked by reading, not compiled, not run on hardware
//
#include <hip/hip_runtime.h>

typedef _Float16 v8h  __attribute__((ext_vector_type(8)));
typedef _Float16 v16h __attribute__((ext_vector_type(16)));
typedef float    v8f  __attribute__((ext_vector_type(8)));
typedef int      v4i  __attribute__((ext_vector_type(4)));

#if defined(__gfx1250__) && __has_builtin(__builtin_amdgcn_global_load_async_to_lds_b128)
#define HAVE_ASYNC_LDS 1
#else
#define HAVE_ASYNC_LDS 0
#endif

// 16-byte global -> LDS copy; async (ASYNCcnt, no VGPR round-trip) when available.
// Builtin signature (from hipcc diagnostic): (int4 AS1* src, int4 AS3* dst, imm, imm)
__device__ inline void copy16_g2l(const _Float16* g, _Float16* l) {
#if HAVE_ASYNC_LDS
  __builtin_amdgcn_global_load_async_to_lds_b128(
      (__attribute__((address_space(1))) v4i*)g,
      (__attribute__((address_space(3))) v4i*)l, 0, 0);
#else
  *(v8h*)l = *(const v8h*)g;
#endif
}

__device__ inline void wait_async_copies() {
#if HAVE_ASYNC_LDS
#if __has_builtin(__builtin_amdgcn_s_wait_asynccnt)
  __builtin_amdgcn_s_wait_asynccnt(0);
#else
  asm volatile("s_wait_asynccnt 0x0" ::: "memory");
#endif
#endif
}

__device__ inline v16h frag16(const _Float16* p0, const _Float16* p1) {
  v8h lo = *(const v8h*)p0;
  v8h hi = *(const v8h*)p1;
  return __builtin_shufflevector(lo, hi, 0,1,2,3,4,5,6,7,8,9,10,11,12,13,14,15);
}

// ---------------------------------------------------------------------------
// Input prep: obs [1024][3][11][11] f32 -> actA [1024][121][64] f16, ch-last,
// channels 3..63 zero (lets the first conv reuse the Cin=64 WMMA kernel).
// ---------------------------------------------------------------------------
__global__ __launch_bounds__(256) void iprep(const float* __restrict__ obs,
                                             _Float16* __restrict__ act) {
  int b = blockIdx.x, tid = threadIdx.x;
  for (int idx = tid; idx < 7744; idx += 256) {
    int p = idx >> 6, c = idx & 63;
    float vv = (c < 3) ? obs[(size_t)(b * 3 + c) * 121 + p] : 0.f;
    act[(size_t)b * 7744 + idx] = (_Float16)vv;
  }
}

// ---------------------------------------------------------------------------
// Weight repack: w [O][Cin][3][3] f32 -> dst [O][ j*64 + c ] f16 (K=576),
// zero-padded channels for Cin<64.
// ---------------------------------------------------------------------------
__global__ __launch_bounds__(576) void wprep(const float* __restrict__ src,
                                             _Float16* __restrict__ dst, int cin) {
  int o = blockIdx.x, t = threadIdx.x;   // t < 576
  int j = t >> 6, c = t & 63;
  float vv = (c < cin) ? src[(o * cin + c) * 9 + j] : 0.f;
  dst[o * 576 + t] = (_Float16)vv;
}

// ---------------------------------------------------------------------------
// 3x3 SAME conv as implicit GEMM via V_WMMA_F32_16X16X32_F16.
// One block = one image. 8 waves = 8 M-tiles (pixels), 4 N-tiles each (64 ch).
// K = 576 in 18 chunks of 32; chunk c -> tap j=c/2, channel base (c&1)*32.
// Staging (image + double-buffered B fragments) uses async global->LDS.
// ---------------------------------------------------------------------------
__global__ __launch_bounds__(256) void conv3x3_wmma(
    const _Float16* __restrict__ in, _Float16* __restrict__ out,
    const _Float16* __restrict__ wgt, const float* __restrict__ bias, int hasres) {
  __shared__ _Float16 img[169 * 64];        // padded 13x13, channel-last
  __shared__ _Float16 bfrag[2][4 * 32 * 16];// double-buffered B fragments

  const int b   = blockIdx.x;
  const int tid = threadIdx.x;
  const int wv  = tid >> 5;   // wave id == M tile
  const int ln  = tid & 31;

  // Prefetch weight block (72KB, L2-resident across 1024 blocks)
  for (int a = tid * 64; a < 36864; a += 256 * 64)
    __builtin_prefetch(wgt + a, 0, 3);

  // Zero padded image (border), then async-copy the 121x64 interior.
  {
    v8h* dst = (v8h*)img;
    v8h z;
#pragma unroll
    for (int r = 0; r < 8; ++r) z[r] = (_Float16)0.f;
    for (int i = tid; i < 169 * 8; i += 256) dst[i] = z;
    __syncthreads();
    const _Float16* src = in + (size_t)b * 7744;
    for (int i = tid; i < 121 * 8; i += 256) {
      int p = i >> 3, g = i & 7;
      int y = p / 11, x = p - y * 11;
      copy16_g2l(src + (size_t)i * 8, &img[(((y + 1) * 13 + (x + 1)) * 8 + g) * 8]);
    }
  }

  // Cooperative async B-fragment staging for chunk c into buffer buf
  auto stageB = [&](int c, int buf) {
    int tile = tid >> 6;
    int rem  = tid & 63;
    int l    = rem >> 1;
    int hb   = (rem & 1) * 8;
    int o    = tile * 16 + (l & 15);
    int k    = c * 32 + ((l < 16) ? 0 : 8) + hb * 2;
    copy16_g2l(&wgt[o * 576 + k], &bfrag[buf][(tile * 32 + l) * 16 + hb]);
  };

  stageB(0, 0);
  wait_async_copies();
  __syncthreads();

  v8f acc[4];
#pragma unroll
  for (int t = 0; t < 4; ++t)
#pragma unroll
    for (int r = 0; r < 8; ++r) acc[t][r] = 0.f;

  // This lane's A row (pixel); wave 7 rows >=121 are clamped (results masked)
  int prow = wv * 16 + (ln & 15);
  if (prow > 120) prow = 120;
  const int py = prow / 11, px = prow - py * 11;
  const int cofs = (ln < 16) ? 0 : 8;

  for (int c = 0; c < 18; ++c) {
    if (c + 1 < 18) stageB(c + 1, (c + 1) & 1);   // async: overlaps the WMMAs below
    int j  = c >> 1;
    int cb = (c & 1) * 32;
    int ky = j / 3, kx = j - ky * 3;
    int base = ((py + ky) * 13 + (px + kx)) * 64 + cb + cofs;
    v16h a = frag16(&img[base], &img[base + 16]);
    const int bb = c & 1;
#pragma unroll
    for (int t = 0; t < 4; ++t) {
      const _Float16* bp = &bfrag[bb][(t * 32 + ln) * 16];
      v16h bf = frag16(bp, bp + 8);
      acc[t] = __builtin_amdgcn_wmma_f32_16x16x32_f16(
          false, a, false, bf, (short)0, acc[t], false, false);
    }
    wait_async_copies();
    __syncthreads();
  }

  // Epilogue: bias (+residual) + ReLU, f16 store, channel-last
  const int pl = (ln < 16) ? 0 : 8;
#pragma unroll
  for (int t = 0; t < 4; ++t) {
    int o = t * 16 + (ln & 15);
    float bo = bias[o];
#pragma unroll
    for (int r = 0; r < 8; ++r) {
      int p = wv * 16 + r + pl;
      if (p < 121) {
        float vv = acc[t][r] + bo;
        size_t oi = (size_t)b * 7744 + (size_t)p * 64 + o;
        if (hasres) vv += (float)out[oi];
        out[oi] = (_Float16)(vv > 0.f ? vv : 0.f);
      }
    }
  }
}

// ---------------------------------------------------------------------------
// Conv1x1(64->16)+ReLU, flatten (ch*121+pix), dot with obs_w -> o4[1024][16]
// ---------------------------------------------------------------------------
__global__ __launch_bounds__(256) void head1(
    const _Float16* __restrict__ act, const float* __restrict__ clw,
    const float* __restrict__ clb, const float* __restrict__ obsw,
    float* __restrict__ o4) {
  __shared__ float feat[1936];
  __shared__ float red[256];
  int b = blockIdx.x, tid = threadIdx.x;
  for (int idx = tid; idx < 1936; idx += 256) {
    int u = idx / 121, p = idx - u * 121;
    float s = clb[u];
    const _Float16* row = act + (size_t)b * 7744 + (size_t)p * 64;
    for (int ci = 0; ci < 64; ++ci) s += (float)row[ci] * clw[u * 64 + ci];
    feat[idx] = s > 0.f ? s : 0.f;
  }
  __syncthreads();
  int uo = tid >> 4, seg = tid & 15;
  float s = 0.f;
  const float* wrow = obsw + uo * 1936 + seg * 121;
  for (int r = 0; r < 121; ++r) s += feat[seg * 121 + r] * wrow[r];
  red[tid] = s;
  __syncthreads();
  if (tid < 16) {
    float t = 0.f;
    for (int k = 0; k < 16; ++k) t += red[tid * 16 + k];
    o4[b * 16 + tid] = t;
  }
}

// ---------------------------------------------------------------------------
// Fused out_proj * W_O:  A2[e][u] = sum_f wo[u][f]*outp_w[f][e]; b2 = wo@outp_b
// ---------------------------------------------------------------------------
__global__ void a2prep(const float* __restrict__ wo, const float* __restrict__ outpw,
                       const float* __restrict__ outpb,
                       float* __restrict__ A2, float* __restrict__ b2) {
  int idx = blockIdx.x * 256 + threadIdx.x;
  if (idx < 4096) {
    int e = idx >> 5, u = idx & 31;
    float s = 0.f;
    for (int f = 0; f < 128; ++f) s += wo[u * 128 + f] * outpw[f * 128 + e];
    A2[idx] = s;
  } else if (idx < 4128) {
    int u = idx - 4096;
    float s = 0.f;
    for (int f = 0; f < 128; ++f) s += wo[u * 128 + f] * outpb[f];
    b2[u] = s;
  }
}

// ---------------------------------------------------------------------------
// c = [mean(o4)+obs_b , act_embed]; q/k/v = (c@W.T+b)@in_w.T+in_b
// ---------------------------------------------------------------------------
__global__ __launch_bounds__(128) void cqkv(
    const float* __restrict__ o4, const float* __restrict__ obsb,
    const int* __restrict__ action, const float* __restrict__ actw,
    const float* __restrict__ actb,
    const float* __restrict__ wq, const float* __restrict__ bqv,
    const float* __restrict__ wk, const float* __restrict__ bkv,
    const float* __restrict__ wv, const float* __restrict__ bvv,
    const float* __restrict__ inqw, const float* __restrict__ inqb,
    const float* __restrict__ inkw, const float* __restrict__ inkb,
    const float* __restrict__ invw, const float* __restrict__ invb,
    float* __restrict__ q, float* __restrict__ k, float* __restrict__ v) {
  __shared__ float cL[32];
  __shared__ float t3[3][128];
  int i = blockIdx.x, t = threadIdx.x;
  if (t < 16) {
    cL[t] = 0.25f * (o4[(i * 4 + 0) * 16 + t] + o4[(i * 4 + 1) * 16 + t] +
                     o4[(i * 4 + 2) * 16 + t] + o4[(i * 4 + 3) * 16 + t]) + obsb[t];
  } else if (t < 32) {
    int u = t - 16;
    cL[t] = actw[u * 5 + action[i]] + actb[u];
  }
  __syncthreads();
  {
    float sq = bqv[t], sk = bkv[t], sv = bvv[t];
    for (int u = 0; u < 32; ++u) {
      float cv = cL[u];
      sq += cv * wq[t * 32 + u]; sk += cv * wk[t * 32 + u]; sv += cv * wv[t * 32 + u];
    }
    t3[0][t] = sq; t3[1][t] = sk; t3[2][t] = sv;
  }
  __syncthreads();
  {
    float sq = inqb[t], sk = inkb[t], sv = invb[t];
    for (int f = 0; f < 128; ++f) {
      sq += t3[0][f] * inqw[t * 128 + f];
      sk += t3[1][f] * inkw[t * 128 + f];
      sv += t3[2][f] * invw[t * 128 + f];
    }
    q[i * 128 + t] = sq; k[i * 128 + t] = sk; v[i * 128 + t] = sv;
  }
}

// ---------------------------------------------------------------------------
// S[h][i][j] = (qh . kh) / sqrt(32)
// ---------------------------------------------------------------------------
__global__ __launch_bounds__(128) void skern(const float* __restrict__ q,
                                             const float* __restrict__ k,
                                             float* __restrict__ S) {
  __shared__ float qL[128];
  int i = blockIdx.x, t = threadIdx.x;
  qL[t] = q[i * 128 + t];
  __syncthreads();
  for (int jj = t; jj < 256; jj += 128) {
    const float* kr = k + jj * 128;
    for (int h = 0; h < 4; ++h) {
      float s = 0.f;
      for (int d = 0; d < 32; ++d) s += qL[h * 32 + d] * kr[h * 32 + d];
      S[h * 65536 + i * 256 + jj] = s * 0.17677669529663689f;
    }
  }
}

// ---------------------------------------------------------------------------
// Per-agent masked attention over its neighbor set + dueling head.
// latent_i = (sum_{q in set} ctx_q) @ A2 + |set| * b2
// ---------------------------------------------------------------------------
__global__ __launch_bounds__(128) void agent_k(
    const int* __restrict__ state, const float* __restrict__ S,
    const float* __restrict__ v, const float* __restrict__ A2,
    const float* __restrict__ b2,
    const float* __restrict__ advw, const float* __restrict__ advb,
    const float* __restrict__ valw, const float* __restrict__ valb,
    float* __restrict__ outp) {
  __shared__ int flag[256];
  __shared__ int list[256];
  __shared__ int cntS;
  __shared__ float ctx[128];
  __shared__ float lat[32];
  __shared__ float Aq[6];
  int i = blockIdx.x, t = threadIdx.x;
  int sx = state[i * 2], sy = state[i * 2 + 1];
  for (int j = t; j < 256; j += 128) {
    int dx = state[j * 2] - sx;     if (dx < 0) dx = -dx;
    int dy = state[j * 2 + 1] - sy; if (dy < 0) dy = -dy;
    flag[j] = (j == i) || (j > i && dx <= 5 && dy <= 5);
  }
  __syncthreads();
  if (t == 0) {
    int c = 0;
    for (int j = 0; j < 256; ++j) if (flag[j]) list[c++] = j;
    cntS = c;
  }
  __syncthreads();
  int cnt = cntS;
  int h = t >> 5;
  const float* Sh = S + h * 65536;
  float acc = 0.f;
  for (int qi = 0; qi < cnt; ++qi) {
    const float* Sr = Sh + list[qi] * 256;
    float mx = -1e30f;
    for (int jj = 0; jj < cnt; ++jj) { float sv = Sr[list[jj]]; if (sv > mx) mx = sv; }
    float se = 0.f;
    for (int jj = 0; jj < cnt; ++jj) se += expf(Sr[list[jj]] - mx);
    float inv = 1.f / se;
    for (int jj = 0; jj < cnt; ++jj)
      acc += expf(Sr[list[jj]] - mx) * inv * v[list[jj] * 128 + t];
  }
  ctx[t] = acc;
  __syncthreads();
  if (t < 32) {
    float s = (float)cnt * b2[t];
    for (int e = 0; e < 128; ++e) s += ctx[e] * A2[e * 32 + t];
    lat[t] = s;
  }
  __syncthreads();
  if (t < 5) {
    float s = advb[t];
    for (int u = 0; u < 32; ++u) s += lat[u] * advw[t * 32 + u];
    Aq[t] = s;
  } else if (t == 5) {
    float s = valb[0];
    for (int u = 0; u < 32; ++u) s += lat[u] * valw[u];
    Aq[5] = s;
  }
  __syncthreads();
  if (t < 5) {
    float mean = 0.2f * (Aq[0] + Aq[1] + Aq[2] + Aq[3] + Aq[4]);
    outp[i * 5 + t] = Aq[5] + Aq[t] - mean;
  }
}

// ---------------------------------------------------------------------------
extern "C" void kernel_launch(void* const* d_in, const int* in_sizes, int n_in,
                              void* d_out, int out_size, void* d_ws, size_t ws_size,
                              hipStream_t stream) {
  const float* obs    = (const float*)d_in[0];
  const int*   action = (const int*)d_in[1];
  const int*   state  = (const int*)d_in[2];
  const float* c0_w  = (const float*)d_in[3];
  const float* c0_b  = (const float*)d_in[4];
  const float* cl_w  = (const float*)d_in[5];
  const float* cl_b  = (const float*)d_in[6];
  const float* obs_w = (const float*)d_in[7];
  const float* obs_b = (const float*)d_in[8];
  const float* act_w = (const float*)d_in[9];
  const float* act_b = (const float*)d_in[10];
  const float* wq = (const float*)d_in[11];
  const float* bq = (const float*)d_in[12];
  const float* wk = (const float*)d_in[13];
  const float* bk = (const float*)d_in[14];
  const float* wv = (const float*)d_in[15];
  const float* bv = (const float*)d_in[16];
  const float* inq_w = (const float*)d_in[17];
  const float* inq_b = (const float*)d_in[18];
  const float* ink_w = (const float*)d_in[19];
  const float* ink_b = (const float*)d_in[20];
  const float* inv_w = (const float*)d_in[21];
  const float* inv_b = (const float*)d_in[22];
  const float* outp_w = (const float*)d_in[23];
  const float* outp_b = (const float*)d_in[24];
  const float* wo    = (const float*)d_in[25];
  const float* val_w = (const float*)d_in[26];
  const float* val_b = (const float*)d_in[27];
  const float* adv_w = (const float*)d_in[28];
  const float* adv_b = (const float*)d_in[29];
  const float* rw[6] = {(const float*)d_in[30], (const float*)d_in[32],
                        (const float*)d_in[34], (const float*)d_in[36],
                        (const float*)d_in[38], (const float*)d_in[40]};
  const float* rb[6] = {(const float*)d_in[31], (const float*)d_in[33],
                        (const float*)d_in[35], (const float*)d_in[37],
                        (const float*)d_in[39], (const float*)d_in[41]};

  char* ws = (char*)d_ws;
  size_t off = 0;
  auto alloc = [&](size_t bytes) -> char* {
    char* p = ws + off;
    off += (bytes + 255) & ~(size_t)255;
    return p;
  };
  _Float16* actA = (_Float16*)alloc((size_t)1024 * 7744 * 2);
  _Float16* actB = (_Float16*)alloc((size_t)1024 * 7744 * 2);
  _Float16* wbuf = (_Float16*)alloc((size_t)7 * 36864 * 2);
  float* o4 = (float*)alloc((size_t)1024 * 16 * 4);
  float* qb = (float*)alloc((size_t)256 * 128 * 4);
  float* kb = (float*)alloc((size_t)256 * 128 * 4);
  float* vb = (float*)alloc((size_t)256 * 128 * 4);
  float* Sb = (float*)alloc((size_t)4 * 256 * 256 * 4);
  float* A2 = (float*)alloc((size_t)4096 * 4);
  float* b2 = (float*)alloc((size_t)32 * 4);

  iprep<<<1024, 256, 0, stream>>>(obs, actA);
  wprep<<<64, 576, 0, stream>>>(c0_w, wbuf + 0 * 36864, 3);
  for (int l = 0; l < 6; ++l)
    wprep<<<64, 576, 0, stream>>>(rw[l], wbuf + (size_t)(l + 1) * 36864, 64);

  conv3x3_wmma<<<1024, 256, 0, stream>>>(actA, actB, wbuf + 0 * 36864, c0_b, 0);
  conv3x3_wmma<<<1024, 256, 0, stream>>>(actB, actA, wbuf + 1 * 36864, rb[0], 0);
  conv3x3_wmma<<<1024, 256, 0, stream>>>(actA, actB, wbuf + 2 * 36864, rb[1], 1);
  conv3x3_wmma<<<1024, 256, 0, stream>>>(actB, actA, wbuf + 3 * 36864, rb[2], 0);
  conv3x3_wmma<<<1024, 256, 0, stream>>>(actA, actB, wbuf + 4 * 36864, rb[3], 1);
  conv3x3_wmma<<<1024, 256, 0, stream>>>(actB, actA, wbuf + 5 * 36864, rb[4], 0);
  conv3x3_wmma<<<1024, 256, 0, stream>>>(actA, actB, wbuf + 6 * 36864, rb[5], 1);

  head1<<<1024, 256, 0, stream>>>(actB, cl_w, cl_b, obs_w, o4);
  a2prep<<<17, 256, 0, stream>>>(wo, outp_w, outp_b, A2, b2);
  cqkv<<<256, 128, 0, stream>>>(o4, obs_b, action, act_w, act_b,
                                wq, bq, wk, bk, wv, bv,
                                inq_w, inq_b, ink_w, ink_b, inv_w, inv_b,
                                qb, kb, vb);
  skern<<<256, 128, 0, stream>>>(qb, kb, Sb);
  agent_k<<<256, 128, 0, stream>>>(state, Sb, vb, A2, b2,
                                   adv_w, adv_b, val_w, val_b, (float*)d_out);
}
